// FactoredLinearEMALayer_18442589569720
// MI455X (gfx1250) — matrix operand order repaired and checked by reading, not compiled, not run on hardware
//
#include <hip/hip_runtime.h>
#include <hip/hip_bf16.h>
#include <stdint.h>

// Problem constants (match the reference)
#define B_  2
#define L_  1024
#define D_  512
#define H_  8
#define N_  4
#define V_  32000
#define K_  64
#define HD_ 64

typedef __attribute__((ext_vector_type(16))) __bf16   v16bf;
typedef __attribute__((ext_vector_type(8)))  __bf16   v8bf;
typedef __attribute__((ext_vector_type(8)))  float    v8f;
typedef __attribute__((ext_vector_type(4)))  uint32_t v4u;
typedef __attribute__((ext_vector_type(8)))  int      v8i;
typedef __attribute__((ext_vector_type(4)))  int      v4i;

__device__ __forceinline__ __bf16 f2bf(float f) {
  uint32_t u = __builtin_bit_cast(uint32_t, f);
  uint32_t r = u + 0x7fffu + ((u >> 16) & 1u);   // round-to-nearest-even
  uint16_t s = (uint16_t)(r >> 16);
  return __builtin_bit_cast(__bf16, s);
}
__device__ __forceinline__ float bf2f(__bf16 h) {
  uint16_t s = __builtin_bit_cast(uint16_t, h);
  uint32_t u = ((uint32_t)s) << 16;
  return __builtin_bit_cast(float, u);
}

// ---------------------------------------------------------------------------
// 1) Row layernorm (D=512) + cast to bf16.  One wave32 per row, 8 rows/block.
// ---------------------------------------------------------------------------
__global__ __launch_bounds__(256)
void ln_rows_bf16(const float* __restrict__ X, const float* __restrict__ g,
                  const float* __restrict__ b, __bf16* __restrict__ Y, int rows)
{
  const int wave = threadIdx.x >> 5, lane = threadIdx.x & 31;
  const int row = blockIdx.x * 8 + wave;
  if (row >= rows) return;
  const float* x = X + (size_t)row * D_;
  float v[16], s = 0.f, s2 = 0.f;
#pragma unroll
  for (int j = 0; j < 16; ++j) {
    float t = x[lane + 32 * j];
    v[j] = t; s += t; s2 += t * t;
  }
#pragma unroll
  for (int o = 16; o > 0; o >>= 1) { s += __shfl_xor(s, o, 32); s2 += __shfl_xor(s2, o, 32); }
  const float mean = s * (1.f / D_);
  const float inv  = rsqrtf(s2 * (1.f / D_) - mean * mean + 1e-5f);
#pragma unroll
  for (int j = 0; j < 16; ++j) {
    int d = lane + 32 * j;
    Y[(size_t)row * D_ + d] = f2bf((v[j] - mean) * inv * g[d] + b[d]);
  }
}

// ---------------------------------------------------------------------------
// 2) P_gate + layernorm + cast: rows = B*L.  G = LN(x_q * ((C+1)@W_pe))
// ---------------------------------------------------------------------------
__global__ __launch_bounds__(256)
void gate_ln_bf16(const float* __restrict__ x_q, const float* __restrict__ C_seq,
                  const float* __restrict__ W_pe, const float* __restrict__ g,
                  const float* __restrict__ b, __bf16* __restrict__ Y)
{
  const int wave = threadIdx.x >> 5, lane = threadIdx.x & 31;
  const int row = blockIdx.x * 8 + wave;          // row in [0, B*L)
  const float* x = x_q + (size_t)row * D_;
  const float c0 = C_seq[row * N_ + 0] + 1.f;
  const float c1 = C_seq[row * N_ + 1] + 1.f;
  const float c2 = C_seq[row * N_ + 2] + 1.f;
  const float c3 = C_seq[row * N_ + 3] + 1.f;
  float v[16], s = 0.f, s2 = 0.f;
#pragma unroll
  for (int j = 0; j < 16; ++j) {
    int d = lane + 32 * j;
    float p = c0 * W_pe[d] + c1 * W_pe[D_ + d] + c2 * W_pe[2 * D_ + d] + c3 * W_pe[3 * D_ + d];
    float t = x[d] * p;
    v[j] = t; s += t; s2 += t * t;
  }
#pragma unroll
  for (int o = 16; o > 0; o >>= 1) { s += __shfl_xor(s, o, 32); s2 += __shfl_xor(s2, o, 32); }
  const float mean = s * (1.f / D_);
  const float inv  = rsqrtf(s2 * (1.f / D_) - mean * mean + 1e-5f);
#pragma unroll
  for (int j = 0; j < 16; ++j) {
    int d = lane + 32 * j;
    Y[(size_t)row * D_ + d] = f2bf((v[j] - mean) * inv * g[d] + b[d]);
  }
}

// ---------------------------------------------------------------------------
// 3) Transpose+cast a 512x512 f32 weight to bf16: WT[n*512+k] = W[k*512+n]
// ---------------------------------------------------------------------------
__global__ __launch_bounds__(256)
void transpose_cast_bf16(const float* __restrict__ W, __bf16* __restrict__ WT)
{
  const int idx = blockIdx.x * 256 + threadIdx.x;  // 512*512 total
  const int n = idx >> 9, k = idx & 511;
  WT[idx] = f2bf(W[(size_t)k * D_ + n]);
}

// ---------------------------------------------------------------------------
// 4) Generic bf16 WMMA GEMM: C(MxD_) = A(MxD_, bf16 row-major) @ B,
//    with B given transposed: BT[n*512+k] (bf16).  f32 accumulate.
//    Block = 16 output rows, 8 waves; wave w owns column tiles 4w..4w+3.
//    k-outer / tile-inner: one A fragment feeds 4 WMMAs (4 accumulators),
//    quartering A-side vmem issue pressure vs. tile-outer.
// ---------------------------------------------------------------------------
template <bool OUT_BF16>
__global__ __launch_bounds__(256)
void gemm_bf16(const __bf16* __restrict__ A, const __bf16* __restrict__ BT,
               void* __restrict__ Cout)
{
  const int lane = threadIdx.x & 31;
  const int wave = threadIdx.x >> 5;
  const int row0 = blockIdx.x * 16;
  const int m    = lane & 15;
  const bool hi  = lane >= 16;
  const int abase = hi ? 16 : 0;          // A: lanes>=16 hold K = 16..31
  const __bf16* arow = A + (size_t)(row0 + m) * D_;

  const __bf16* bcol[4];
#pragma unroll
  for (int c = 0; c < 4; ++c)
    bcol[c] = BT + (size_t)((wave * 4 + c) * 16 + (lane & 15)) * D_ + (hi ? 16 : 0);

  v8f acc[4] = {};
  for (int k0 = 0; k0 < D_; k0 += 32) {
    __builtin_prefetch((const void*)(arow + k0 + 64), 0, 3);
    v8bf alo = *(const v8bf*)(arow + k0 + abase);
    v8bf ahi = *(const v8bf*)(arow + k0 + abase + 8);
    v16bf a;
#pragma unroll
    for (int j = 0; j < 8; ++j) { a[j] = alo[j]; a[8 + j] = ahi[j]; }
#pragma unroll
    for (int c = 0; c < 4; ++c) {
      v16bf bb = *(const v16bf*)(bcol[c] + k0);
      acc[c] = __builtin_amdgcn_wmma_f32_16x16x32_bf16(
          false, a, false, bb, (short)0, acc[c], false, false);
    }
  }
#pragma unroll
  for (int c = 0; c < 4; ++c) {
    const int n = (wave * 4 + c) * 16 + (lane & 15);
    if (OUT_BF16) {
      __bf16* C = (__bf16*)Cout;
#pragma unroll
      for (int r = 0; r < 8; ++r)
        C[(size_t)(row0 + r + (hi ? 8 : 0)) * D_ + n] = f2bf(acc[c][r]);
    } else {
      float* C = (float*)Cout;
#pragma unroll
      for (int r = 0; r < 8; ++r)
        C[(size_t)(row0 + r + (hi ? 8 : 0)) * D_ + n] = acc[c][r];
    }
  }
}

// ---------------------------------------------------------------------------
// TDM helper: issue a 1-D tile load (16 x 8B = 128 bytes) into LDS via the
// Tensor Data Mover.  Descriptor layout per CDNA5 ISA ch.8 (async_tensor):
//   group0: [1:0]=count=1, [63:32]=lds_addr, [120:64]=global_addr, [127:126]=2
//   group1: [17:16]=data_size(3->8B), [79:48]=tensor_dim0=16,
//           [111:80]=tensor_dim1=1, [127:112]=tile_dim0=16,
//           [143:128]=tile_dim1=1, [207:160]=tensor_dim0_stride=16
// workgroup_mask=0 (not in a cluster).  Tracked by TENSORcnt.
// ---------------------------------------------------------------------------
__device__ __forceinline__ void tdm_load_row128(uint32_t lds_off, const void* gptr)
{
  uint64_t ga = (uint64_t)(uintptr_t)gptr;
  v4u g0;
  g0[0] = 1u;                                  // count=1
  g0[1] = lds_off;                             // lds_addr
  g0[2] = (uint32_t)ga;                        // global_addr[31:0]
  g0[3] = (uint32_t)(ga >> 32) | (2u << 30);   // global_addr[56:32] | type=2
  v8i g1;
  g1[0] = (3 << 16);                           // data_size = 8B
  g1[1] = (16 << 16);                          // tensor_dim0 = 16
  g1[2] = (1 << 16);                           // tensor_dim1 = 1
  g1[3] = (16 << 16);                          // tile_dim0 = 16
  g1[4] = 1;                                   // tile_dim1 = 1
  g1[5] = 16;                                  // tensor_dim0_stride = 16
  g1[6] = 0;
  g1[7] = 0;
  v4i gz = {0, 0, 0, 0};
#if defined(__clang_major__) && (__clang_major__ >= 23)
  v8i gz8 = {0, 0, 0, 0, 0, 0, 0, 0};
  __builtin_amdgcn_tensor_load_to_lds(g0, g1, gz, gz, gz8, 0);
#else
  __builtin_amdgcn_tensor_load_to_lds(g0, g1, gz, gz, 0);
#endif
}

// ---------------------------------------------------------------------------
// 5) Fused S0 + EMA scan. One workgroup per (b,h,n): 64 WGs, 256 threads.
//    State S_n is 64x64 f32 in REGISTERS: thread t owns row i=t>>2,
//    cols j0..j0+15, j0=(t&3)*16.  Per-step E_k/E_v rows (128B each) are
//    staged LDS-direct by the Tensor Data Mover (wave 0 issues, waits on
//    TENSORcnt, then the workgroup barrier publishes them).
// ---------------------------------------------------------------------------
__global__ __launch_bounds__(256)
void s0_scan(const __bf16* __restrict__ EK, const __bf16* __restrict__ EV,
             const float* __restrict__ Qf, const int* __restrict__ xtok,
             const float* __restrict__ rhos, const float* __restrict__ c_base,
             const int* __restrict__ uslots, const unsigned char* __restrict__ pmask,
             const float* __restrict__ W_pe, float* __restrict__ Opart)
{
  __shared__ __bf16 ls_ek[K_ * HD_];
  __shared__ __bf16 ls_ev[K_ * HD_];
  __shared__ int    ls_slot[K_];
  __shared__ float  ls_c[K_];
  __shared__ float  ls_q[HD_];
  __shared__ __bf16 ls_ekb[HD_];     // TDM destination (128 B)
  __shared__ __bf16 ls_evb[HD_];     // TDM destination (128 B)
  __shared__ float  ls_op[8 * HD_];

  const int bid = blockIdx.x;            // = b*32 + h*4 + n
  const int b = bid >> 5;
  const int h = (bid >> 2) & 7;
  const int n = bid & 3;
  const int t = threadIdx.x;
  const int i  = t >> 2;                 // state row owned by this thread
  const int j0 = (t & 3) << 4;           // first of 16 owned cols
  const float rho = rhos[n];
  const float wpe_i = W_pe[n * D_ + h * HD_ + i];

  const uint32_t ek_off = (uint32_t)(uintptr_t)(void*)ls_ekb;  // LDS byte offset
  const uint32_t ev_off = (uint32_t)(uintptr_t)(void*)ls_evb;

  // ---- stage the K=64 base slots ----
  if (t < K_) {
    int s = uslots[b * K_ + t];
    ls_slot[t] = s < 0 ? 0 : s;
    ls_c[t] = pmask[b * K_ + t] ? c_base[(b * K_ + t) * N_ + n] : 0.f;
  }
  __syncthreads();
  {
    const int k = t >> 2, part = t & 3;
    const size_t src = (size_t)ls_slot[k] * D_ + h * HD_ + part * 16;
    *(v8bf*)&ls_ek[k * HD_ + part * 16]     = *(const v8bf*)(EK + src);
    *(v8bf*)&ls_ek[k * HD_ + part * 16 + 8] = *(const v8bf*)(EK + src + 8);
    *(v8bf*)&ls_ev[k * HD_ + part * 16]     = *(const v8bf*)(EV + src);
    *(v8bf*)&ls_ev[k * HD_ + part * 16 + 8] = *(const v8bf*)(EV + src + 8);
  }
  __syncthreads();

  // ---- S0 = sum_k c[k] * ek[k] (outer) ev[k], this thread's 16 elements ----
  float S[16];
#pragma unroll
  for (int jj = 0; jj < 16; ++jj) S[jj] = 0.f;
  for (int k = 0; k < K_; ++k) {
    const float cek = ls_c[k] * bf2f(ls_ek[k * HD_ + i]);
#pragma unroll
    for (int jj = 0; jj < 16; ++jj)
      S[jj] += cek * bf2f(ls_ev[k * HD_ + j0 + jj]);
  }
  __syncthreads();

  // ---- scan over L tokens ----
  for (int ts = 0; ts < L_; ++ts) {
    const int tok = xtok[b * L_ + ts];   // wave-uniform (scalarized)
    if (t < 64) {
      ls_q[t] = Qf[((size_t)(b * L_ + ts)) * D_ + h * HD_ + t];
    }
    if (t < 32) {
      // wave 0 drives the TDM: two 128B rows, then drain TENSORcnt
      tdm_load_row128(ek_off, (const void*)(EK + (size_t)tok * D_ + h * HD_));
      tdm_load_row128(ev_off, (const void*)(EV + (size_t)tok * D_ + h * HD_));
      __builtin_amdgcn_s_wait_tensorcnt(0);
    }
    __syncthreads();

    const float qm  = ls_q[i] * wpe_i;     // q_mod for this state row
    const float eki = bf2f(ls_ekb[i]);
#pragma unroll
    for (int jj = 0; jj < 16; ++jj) {
      S[jj] *= rho;                        // decay
      float pv = qm * S[jj];               // o_t contribution (post-decay, pre-update)
      pv += __shfl_xor(pv, 4, 32);
      pv += __shfl_xor(pv, 8, 32);
      pv += __shfl_xor(pv, 16, 32);        // sum over the 8 state rows in this wave
      if ((t & 31) < 4) ls_op[(t >> 5) * HD_ + j0 + jj] = pv;
      S[jj] += eki * bf2f(ls_evb[j0 + jj]); // rank-1 update
    }
    __syncthreads();

    if (t < 64) {
      float o = 0.f;
#pragma unroll
      for (int w = 0; w < 8; ++w) o += ls_op[w * HD_ + t];
      Opart[((size_t)bid * L_ + ts) * HD_ + t] = o;
    }
    // next iteration's __syncthreads() separates reduce-reads from restaging
  }
}

// ---------------------------------------------------------------------------
// 6) Merge partial outputs over n, cast to bf16: Obf[(b*L+l)*D + h*64+e]
// ---------------------------------------------------------------------------
__global__ __launch_bounds__(256)
void merge_cast(const float* __restrict__ Opart, __bf16* __restrict__ Obf)
{
  const int idx = blockIdx.x * 256 + threadIdx.x;   // B*L*D total
  const int row = idx >> 9;                         // b*L + l
  const int d   = idx & 511;
  const int h = d >> 6, e = d & 63;
  const int b = row >> 10, l = row & 1023;
  float s = 0.f;
#pragma unroll
  for (int n = 0; n < N_; ++n)
    s += Opart[((size_t)(b * 32 + h * 4 + n) * L_ + l) * HD_ + e];
  Obf[idx] = f2bf(s);
}

// ---------------------------------------------------------------------------
// Host-side launch
// ---------------------------------------------------------------------------
extern "C" void kernel_launch(void* const* d_in, const int* in_sizes, int n_in,
                              void* d_out, int out_size, void* d_ws, size_t ws_size,
                              hipStream_t stream) {
  (void)in_sizes; (void)n_in; (void)out_size; (void)ws_size;
  const float* x_q    = (const float*)d_in[0];
  const int*   x      = (const int*)d_in[1];
  const float* Eslots = (const float*)d_in[2];
  const float* rhos   = (const float*)d_in[3];
  const float* C_seq  = (const float*)d_in[4];
  const float* c_base = (const float*)d_in[5];
  const int*   uslots = (const int*)d_in[6];
  const unsigned char* pmask = (const unsigned char*)d_in[7];
  const float* Wq = (const float*)d_in[8];
  const float* Wk = (const float*)d_in[9];
  const float* Wv = (const float*)d_in[10];
  const float* Wo = (const float*)d_in[11];
  const float* W_pe = (const float*)d_in[12];
  const float* ln_kv_g = (const float*)d_in[13];
  const float* ln_kv_b = (const float*)d_in[14];
  const float* ln_q_g  = (const float*)d_in[15];
  const float* ln_q_b  = (const float*)d_in[16];

  // workspace carve-up (256B aligned)
  char* base = (char*)d_ws;
  size_t off = 0;
  auto alloc = [&](size_t bytes) { size_t o = off; off += (bytes + 255) & ~(size_t)255; return o; };
  __bf16* MB  = (__bf16*)(base + alloc((size_t)V_ * D_ * 2));       // layernormed E_slots
  __bf16* EK  = (__bf16*)(base + alloc((size_t)V_ * D_ * 2));       // E_k (bf16)
  __bf16* EV  = (__bf16*)(base + alloc((size_t)V_ * D_ * 2));       // E_v (bf16)
  __bf16* WkT = (__bf16*)(base + alloc((size_t)D_ * D_ * 2));
  __bf16* WvT = (__bf16*)(base + alloc((size_t)D_ * D_ * 2));
  __bf16* WqT = (__bf16*)(base + alloc((size_t)D_ * D_ * 2));
  __bf16* WoT = (__bf16*)(base + alloc((size_t)D_ * D_ * 2));
  __bf16* Qg  = (__bf16*)(base + alloc((size_t)B_ * L_ * D_ * 2));  // gated+LN input to Wq
  float*  Qf  = (float*)(base + alloc((size_t)B_ * L_ * D_ * 4));   // Q (f32)
  float*  Opart = (float*)(base + alloc((size_t)B_ * H_ * N_ * L_ * HD_ * 4));
  __bf16* Obf = (__bf16*)(base + alloc((size_t)B_ * L_ * D_ * 2));

  // 1) layernorm E_slots -> MB (bf16)
  ln_rows_bf16<<<V_ / 8, 256, 0, stream>>>(Eslots, ln_kv_g, ln_kv_b, MB, V_);

  // 2) transpose+cast weights
  transpose_cast_bf16<<<(D_ * D_) / 256, 256, 0, stream>>>(Wk, WkT);
  transpose_cast_bf16<<<(D_ * D_) / 256, 256, 0, stream>>>(Wv, WvT);
  transpose_cast_bf16<<<(D_ * D_) / 256, 256, 0, stream>>>(Wq, WqT);
  transpose_cast_bf16<<<(D_ * D_) / 256, 256, 0, stream>>>(Wo, WoT);

  // 3) big WMMA GEMMs: E_k, E_v  (compute-bound; bf16 WMMA path)
  gemm_bf16<true><<<V_ / 16, 256, 0, stream>>>(MB, WkT, (void*)EK);
  gemm_bf16<true><<<V_ / 16, 256, 0, stream>>>(MB, WvT, (void*)EV);

  // 4) gate + LN, then Q = Qg @ Wq (f32 out)
  gate_ln_bf16<<<(B_ * L_) / 8, 256, 0, stream>>>(x_q, C_seq, W_pe, ln_q_g, ln_q_b, Qg);
  gemm_bf16<false><<<(B_ * L_) / 16, 256, 0, stream>>>(Qg, WqT, (void*)Qf);

  // 5) fused S0 + scan: one WG per (b,h,n)
  s0_scan<<<B_ * H_ * N_, 256, 0, stream>>>(EK, EV, Qf, x, rhos, c_base,
                                            uslots, pmask, W_pe, Opart);

  // 6) merge over n, cast bf16
  merge_cast<<<(B_ * L_ * D_) / 256, 256, 0, stream>>>(Opart, Obf);

  // 7) final projection -> d_out (f32)
  gemm_bf16<false><<<(B_ * L_) / 16, 256, 0, stream>>>(Obf, WoT, d_out);
}
